// BaumWelch_76879914598444
// MI455X (gfx1250) — matrix operand order repaired and checked by reading, not compiled, not run on hardware
//
#include <hip/hip_runtime.h>
#include <math.h>

// ---------------------------------------------------------------------------
// HMM forward-backward (Baum-Welch gamma) for Nb=256, T=512, K=256, V=4096
// on MI455X (gfx1250, wave32, WMMA).
//
// Per recursion step (per direction) the logsumexp-matvec is computed as a
// max-shifted GEMM  S = u @ expA  using v_wmma_f32_16x16x32_bf16 with the
// bf16 expA matrix held entirely in VGPRs and the A-operand split into
// bf16 hi+lo for error compensation (f32 accumulate).
// ---------------------------------------------------------------------------

typedef __bf16 bf16;
typedef __attribute__((ext_vector_type(8)))  bf16  v8bf;
typedef __attribute__((ext_vector_type(16))) bf16  v16bf;
typedef __attribute__((ext_vector_type(8)))  float v8f;

#define NB 256
#define TT 512
#define KK 256
#define VV 4096
#define AP 264   // aprev row stride (floats)  -> bank-conflict-free
#define US 264   // uh/ul row stride (bf16)    -> 528B rows, conflict-free b128

union SharedU {
  bf16 stage[KK * KK];                 // 128 KB: bf16 exp(A) staging (prologue)
  struct {
    float aprev[16 * AP];              // alpha_prev (fwd) / beta+b work (bwd)
    bf16  uh[16 * US];                 // hi bf16 split of u = exp(a - m)
    bf16  ul[16 * US];                 // lo bf16 split
    float m[16];                       // per-sequence row max
  } run;
};

__device__ inline v16bf ld_b16_contig(const bf16* p) {
  const v8bf* q = (const v8bf*)p;
  v8bf a = q[0], b = q[1];
  return __builtin_shufflevector(a, b, 0,1,2,3,4,5,6,7,8,9,10,11,12,13,14,15);
}

// A-operand (16-bit A 16x32 layout): elements 0..7 at p[0..7], 8..15 at p[16..23]
__device__ inline v16bf ld_a_op(const bf16* p) {
  const v8bf* q0 = (const v8bf*)p;
  const v8bf* q1 = (const v8bf*)(p + 16);
  v8bf a = *q0, b = *q1;
  return __builtin_shufflevector(a, b, 0,1,2,3,4,5,6,7,8,9,10,11,12,13,14,15);
}

__device__ inline v8f wmma_bf16(v16bf a, v16bf b, v8f c) {
  return __builtin_amdgcn_wmma_f32_16x16x32_bf16(false, a, false, b,
                                                 (short)0, c, false, false);
}

// ---------------------------------------------------------------------------
// Kernel 1: gather logB_obs[n][t][j] = log_B[j][obs[n][t]]
// ---------------------------------------------------------------------------
__global__ __launch_bounds__(256)
void gather_logB(const float* __restrict__ log_B,
                 const int* __restrict__ obs,
                 float* __restrict__ out) {
  const int nt = blockIdx.x;                 // 0 .. Nb*T-1
  const int v  = obs[nt];                    // uniform per block
  const int j  = threadIdx.x;
  out[(size_t)nt * KK + j] = log_B[(size_t)j * VV + v];
}

// ---------------------------------------------------------------------------
// Kernel 2: 16 forward WGs (alpha -> d_out) + 16 backward WGs (beta -> ws),
//           each WG owns 16 sequences, 256 threads = 8 wave32.
// ---------------------------------------------------------------------------
__global__ __launch_bounds__(256, 1)
void hmm_recursion(const float* __restrict__ log_pi,
                   const float* __restrict__ log_A,
                   const float* __restrict__ logBo,
                   float* __restrict__ alpha,   // d_out (holds alpha for now)
                   float* __restrict__ beta) {  // ws beta buffer
  __shared__ SharedU lds;

  const int tid   = threadIdx.x;
  const int lane  = tid & 31;
  const int wid   = tid >> 5;           // wave 0..7
  const int mrow  = lane & 15;
  const int half  = lane >> 4;
  const bool isFwd = (blockIdx.x < 16);
  const int seq0   = (isFwd ? blockIdx.x : (blockIdx.x - 16)) * 16;

  // ---- Prologue A: stage bf16 exp(log_A); transposed for forward ----------
  for (int idx = tid; idx < KK * KK; idx += 256) {
    const int i = idx >> 8, j = idx & 255;
    const float e = __expf(log_A[idx]);
    const int dst = isFwd ? (j * KK + i) : idx;
    lds.stage[dst] = (bf16)e;
  }
  __syncthreads();

  // ---- Prologue B: preload whole B-matrix (expA) into VGPRs ---------------
  // wave w owns output-state tiles 2w, 2w+1 (N = states, 16 each).
  v16bf Bop[2][8];
#pragma unroll
  for (int tt = 0; tt < 2; ++tt) {
    const int j = (wid * 2 + tt) * 16 + mrow;   // B-matrix: lane holds column N=j
#pragma unroll
    for (int c = 0; c < 8; ++c) {
      const int off = c * 32 + half * 8;        // K range of this lane-half
      Bop[tt][c] = ld_a_op(&lds.stage[j * KK + off]);
    }
  }
  __syncthreads();

  // ---- Prologue C: initialize state arrays (union now owns lds.run) -------
  {
    const int n  = tid >> 4;
    const int c0 = (tid & 15) * 16;
#pragma unroll
    for (int k = 0; k < 16; ++k) {
      const int j = c0 + k;
      if (isFwd) {
        const size_t g = ((size_t)(seq0 + n) * TT + 0) * KK + j;
        const float val = log_pi[j] + logBo[g];
        lds.run.aprev[n * AP + j] = val;
        alpha[g] = val;
      } else {
        const size_t g = ((size_t)(seq0 + n) * TT + (TT - 1)) * KK + j;
        lds.run.aprev[n * AP + j] = logBo[g];   // work = beta_{T-1}(=0) + b_{T-1}
        beta[g] = 0.0f;
      }
    }
  }

  // ---- Recursion ----------------------------------------------------------
  for (int s = 1; s < TT; ++s) {
    const int t = isFwd ? s : (TT - 1 - s);
    __syncthreads();                               // state from previous step

    // Phase 2: row maxima m[n]
    {
      const int n = tid >> 4, c = tid & 15;
      const float* row = &lds.run.aprev[n * AP + c * 16];
      float mx = row[0];
#pragma unroll
      for (int k = 1; k < 16; ++k) mx = fmaxf(mx, row[k]);
#pragma unroll
      for (int o = 8; o >= 1; o >>= 1) mx = fmaxf(mx, __shfl_xor(mx, o, 16));
      if (c == 0) lds.run.m[n] = mx;
    }
    __syncthreads();

    // Phase 3: u = exp(a - m), split into bf16 hi + lo, packed b32 stores
    {
      const int n = tid >> 4, c = tid & 15;
      const float mn = lds.run.m[n];
      const float* row = &lds.run.aprev[n * AP + c * 16];
      unsigned* uhrow = (unsigned*)&lds.run.uh[n * US] + c * 8;
      unsigned* ulrow = (unsigned*)&lds.run.ul[n * US] + c * 8;
#pragma unroll
      for (int k2 = 0; k2 < 8; ++k2) {
        const float e0 = __expf(row[2 * k2 + 0] - mn);
        const float e1 = __expf(row[2 * k2 + 1] - mn);
        const bf16 h0 = (bf16)e0, h1 = (bf16)e1;
        const bf16 l0 = (bf16)(e0 - (float)h0), l1 = (bf16)(e1 - (float)h1);
        union { bf16 b[2]; unsigned u; } ph, pl;
        ph.b[0] = h0; ph.b[1] = h1;
        pl.b[0] = l0; pl.b[1] = l1;
        uhrow[k2] = ph.u;
        ulrow[k2] = pl.u;
      }
    }
    __syncthreads();

    // Phase 4: S = (u_hi + u_lo) @ expA via WMMA, f32 accumulate
    v8f acc0 = {0.f,0.f,0.f,0.f,0.f,0.f,0.f,0.f};
    v8f acc1 = {0.f,0.f,0.f,0.f,0.f,0.f,0.f,0.f};
    {
      const bf16* uh = lds.run.uh + mrow * US;
      const bf16* ul = lds.run.ul + mrow * US;
#pragma unroll
      for (int c = 0; c < 8; ++c) {
        const int kb = c * 32 + half * 8;
        const v16bf aH = ld_a_op(uh + kb);
        const v16bf aL = ld_a_op(ul + kb);
        acc0 = wmma_bf16(aH, Bop[0][c], acc0);
        acc1 = wmma_bf16(aH, Bop[1][c], acc1);
        acc0 = wmma_bf16(aL, Bop[0][c], acc0);
        acc1 = wmma_bf16(aL, Bop[1][c], acc1);
      }
    }

    // Phase 5: finish step, write global + next-step LDS state
    {
#pragma unroll
      for (int tt = 0; tt < 2; ++tt) {
        const int j = (wid * 2 + tt) * 16 + mrow;
#pragma unroll
        for (int r = 0; r < 8; ++r) {
          const int n = half * 8 + r;
          const float mn = lds.run.m[n];
          const size_t g = ((size_t)(seq0 + n) * TT + (size_t)t) * KK + j;
          const float sv = (tt == 0) ? acc0[r] : acc1[r];
          const float lg = mn + __logf(sv);
          const float b  = logBo[g];
          if (isFwd) {
            const float val = b + lg;          // alpha_t
            alpha[g] = val;
            lds.run.aprev[n * AP + j] = val;
          } else {
            beta[g] = lg;                      // beta_t
            lds.run.aprev[n * AP + j] = lg + b; // work_t = beta_t + b_t
          }
        }
      }
    }
  }
}

// ---------------------------------------------------------------------------
// Kernel 3: gamma = (alpha + beta) - logsumexp_K(alpha + beta)
//           one wave32 per (n,t) row of K=256.
// ---------------------------------------------------------------------------
__global__ __launch_bounds__(256)
void gamma_combine(const float* __restrict__ beta,
                   float* __restrict__ out) {       // alpha in, gamma out
  const int row  = blockIdx.x * 8 + (threadIdx.x >> 5);
  const int lane = threadIdx.x & 31;
  const float* a = out  + (size_t)row * KK;
  const float* b = beta + (size_t)row * KK;

  float s[8];
  float mx = -__builtin_inff();
#pragma unroll
  for (int k = 0; k < 8; ++k) {
    s[k] = a[lane * 8 + k] + b[lane * 8 + k];
    mx = fmaxf(mx, s[k]);
  }
#pragma unroll
  for (int o = 16; o >= 1; o >>= 1) mx = fmaxf(mx, __shfl_xor(mx, o, 32));
  float sum = 0.f;
#pragma unroll
  for (int k = 0; k < 8; ++k) sum += __expf(s[k] - mx);
#pragma unroll
  for (int o = 16; o >= 1; o >>= 1) sum += __shfl_xor(sum, o, 32);
  const float lse = mx + __logf(sum);

  float* g = out + (size_t)row * KK;
#pragma unroll
  for (int k = 0; k < 8; ++k) g[lane * 8 + k] = s[k] - lse;
}

// ---------------------------------------------------------------------------
extern "C" void kernel_launch(void* const* d_in, const int* in_sizes, int n_in,
                              void* d_out, int out_size, void* d_ws, size_t ws_size,
                              hipStream_t stream) {
  (void)in_sizes; (void)n_in; (void)out_size; (void)ws_size;
  const float* log_pi = (const float*)d_in[0];
  const float* log_A  = (const float*)d_in[1];
  const float* log_B  = (const float*)d_in[2];
  const int*   obs    = (const int*)d_in[3];
  float* out = (float*)d_out;

  float* logBo = (float*)d_ws;                         // Nb*T*K f32 (134 MB)
  float* beta  = logBo + (size_t)NB * TT * KK;         // Nb*T*K f32 (134 MB)

  // 1) emission gather (bandwidth-bound, L2-resident log_B)
  gather_logB<<<NB * TT, 256, 0, stream>>>(log_B, obs, logBo);
  // 2) forward + backward recursions concurrently (16 WGs each)
  hmm_recursion<<<32, 256, 0, stream>>>(log_pi, log_A, logBo, out, beta);
  // 3) gamma normalization (bandwidth-bound)
  gamma_combine<<<(NB * TT) / 8, 256, 0, stream>>>(beta, out);
}
